// RSSM_936302871043
// MI455X (gfx1250) — compile-verified, matrix operand order checked by reading, not run on hardware
//
#include <hip/hip_runtime.h>
#include <hip/hip_bf16.h>
#include <math.h>

// ---------------------------------------------------------------------------
// RSSM scan for MI455X (gfx1250, wave32, WMMA).
// Weight-streaming bandwidth bound at M=16: weights packed once to bf16 WMMA
// B-tiles in d_ws (halves HBM traffic; w_gru fits 192MB L2). Per step: 4 GEMMs
// via v_wmma_f32_16x16x32_bf16 with TDM/async-LDS staged A operand, small
// wave32 kernels for LN / gates / sampling.
// ---------------------------------------------------------------------------

typedef __attribute__((ext_vector_type(16))) __bf16 v16bf;
typedef __attribute__((ext_vector_type(8)))  float  v8f;
typedef unsigned int u32x4 __attribute__((ext_vector_type(4)));
typedef int          i32x4 __attribute__((ext_vector_type(4)));
typedef int          i32x8 __attribute__((ext_vector_type(8)));
typedef int          gv4i  __attribute__((vector_size(16)));   // matches async builtin param

#if defined(__has_builtin)
# if __has_builtin(__builtin_amdgcn_global_load_async_to_lds_b128) && \
     __has_builtin(__builtin_amdgcn_s_wait_asynccnt)
#  define HAVE_ASYNC_LDS 1
# endif
# if __has_builtin(__builtin_amdgcn_tensor_load_to_lds) && \
     __has_builtin(__builtin_amdgcn_s_wait_tensorcnt)
#  define HAVE_TDM 1
# endif
#endif

#define BB      16
#define LL      64
#define DETER_D 4096
#define SC      1024   // STOCH*CLASSES
#define ADIM    6
#define TOKD    4096

#define KT_IN   33     // K=1030 padded to 1056 = 33*32
#define KT_GRU  160    // K=5120
#define KT_OBS  256    // K=8192
#define KT_POST 32     // K=1024

// DMA engines (TDM / async copy) write LDS behind the compiler's back. The
// staging pointer must ESCAPE into the asm (operand, not just a "memory"
// clobber) so alias analysis assumes the asm may store through it; otherwise
// a never-visibly-stored LDS array folds all its loads to undef.
__device__ __forceinline__ void dma_publish(const void* p) {
  asm volatile("" : : "v"(p) : "memory");
}

__device__ __forceinline__ unsigned short f2bf(float f) {
  unsigned u = __float_as_uint(f);
  unsigned r = u + 0x7FFFu + ((u >> 16) & 1u);   // round-to-nearest-even
  return (unsigned short)(r >> 16);
}

__device__ __forceinline__ unsigned hash4(unsigned a, unsigned b, unsigned c, unsigned d) {
  unsigned x = a * 0x9E3779B9u ^ b * 0x85EBCA6Bu ^ c * 0xC2B2AE35u ^ d * 0x27D4EB2Fu;
  x ^= x >> 16; x *= 0x7FEB352Du; x ^= x >> 15; x *= 0x846CA68Bu; x ^= x >> 16;
  return x;
}

#if defined(HAVE_ASYNC_LDS)
// 16-byte async global->LDS copy (ASYNCcnt path). Param type per compiler
// diagnostic: generic pointer to int vector_size(16); integer round-trip
// produces an exact-match pointer and strips const.
__device__ __forceinline__ void async_cp16(const void* gsrc, void* ldst) {
  __builtin_amdgcn_global_load_async_to_lds_b128(
      (gv4i*)(unsigned long long)gsrc, (gv4i*)(unsigned long long)ldst, 0, 0);
}
#endif

#if defined(HAVE_TDM)
// Issue one TDM 2D tile load: global (elem 2B, row stride strideElems) ->
// LDS offset 0, tile tileW x tileH elements.  D# fields per CDNA5 ISA ch.8.
__device__ __forceinline__ void tdm_load_tile_2d(const void* gsrc,
                                                 unsigned strideElems,
                                                 unsigned tensorW,
                                                 unsigned tileW, unsigned tileH) {
  unsigned long long ga = (unsigned long long)gsrc;
  u32x4 g0;
  g0[0] = 1u;                                            // count=1 (valid user D#)
  g0[1] = 0u;                                            // lds_addr = 0 (staging buf at LDS base)
  g0[2] = (unsigned)ga;                                  // global_addr[31:0]
  g0[3] = (unsigned)((ga >> 32) & 0x1FFFFFFu) | (2u << 30);  // addr[56:32] | type=2
  i32x8 g1;
  g1[0] = 0x10000;                                       // data_size=1 (2 bytes/elem)
  g1[1] = (int)((tensorW & 0xFFFFu) << 16);              // tensor_dim0[15:0]
  g1[2] = (int)((tensorW >> 16) | (tileH << 16));        // tensor_dim0[31:16] | tensor_dim1
  g1[3] = (int)(tileW << 16);                            // tile_dim0
  g1[4] = (int)tileH;                                    // tile_dim1 (tile_dim2 = 0)
  g1[5] = (int)strideElems;                              // tensor_dim0_stride[31:0]
  g1[6] = 0;
  g1[7] = 0;
  i32x4 z4 = {0, 0, 0, 0};
#if __clang_major__ >= 23
  i32x8 z8 = {0, 0, 0, 0, 0, 0, 0, 0};
  __builtin_amdgcn_tensor_load_to_lds(g0, g1, z4, z4, z8, 0);
#else
  __builtin_amdgcn_tensor_load_to_lds(g0, g1, z4, z4, 0);
#endif
}
#endif

// --------------------------- weight packing --------------------------------
__global__ void __launch_bounds__(256) pack_weight(
    const float* __restrict__ W, unsigned short* __restrict__ Wp,
    int K, int Kt, int N)
{
  size_t idx = (size_t)blockIdx.x * 256 + threadIdx.x;
  size_t total = (size_t)Kt * (size_t)(N / 16) * 512;
  if (idx >= total) return;
  size_t tile = idx >> 9;
  int e    = (int)(idx & 511);
  int lane = e >> 4, j = e & 15;
  int nt = (int)(tile / (size_t)Kt);
  int kt = (int)(tile % (size_t)Kt);
  int krow = kt * 32 + (lane >> 4) * 16 + j;
  int col  = nt * 16 + (lane & 15);
  float f = (krow < K) ? W[(size_t)krow * N + col] : 0.f;
  Wp[idx] = f2bf(f);
}

// ------------------------------- GEMM --------------------------------------
// Y[16 x N] = X[16 x Kt*32] (bf16) * Wp (packed bf16 32x16 tiles) + bias.
// 256 threads = 8 waves; block covers 256 cols. A chunk staged into LDS by
// the Tensor Data Mover (one wave-level descriptor per chunk) and shared by
// all waves via ds_load_b128; B streams from global as b128 pairs.
__global__ void __launch_bounds__(256) gemm_bf16_wmma(
    const unsigned short* __restrict__ Wp,
    const unsigned short* __restrict__ X,
    const float* __restrict__ bias,
    float* __restrict__ Y,
    int Kt, int N, int ldY)
{
  __shared__ __align__(32) unsigned short lds[16 * 256];
  const int tid  = threadIdx.x;
  const int wave = tid >> 5;
  const int lane = tid & 31;
  const int nt0  = blockIdx.x * 16 + wave * 2;
  const size_t Ktot = (size_t)Kt * 32;

  v8f acc0 = {}, acc1 = {};
  const int nchunk = (Kt + 7) >> 3;
  const int srow = tid >> 4;   // 0..15
  const int sseg = tid & 15;   // 16 segments of 16 bf16 cover 256 cols
  (void)srow; (void)sseg;

  for (int c = 0; c < nchunk; ++c) {
    const int ktBase = c << 3;
    const int ktn = (Kt - ktBase) < 8 ? (Kt - ktBase) : 8;

#if defined(HAVE_TDM)
    if (wave == 0) {                     // wave-level DMA: 16 x 256 bf16 tile
      tdm_load_tile_2d(X + (size_t)ktBase * 32, (unsigned)Ktot,
                       (unsigned)Ktot, 256u, 16u);
      __builtin_amdgcn_s_wait_tensorcnt(0);
    }
#elif defined(HAVE_ASYNC_LDS)
    if (sseg < ktn * 2) {
      const unsigned short* src = X + (size_t)srow * Ktot + (size_t)ktBase * 32 + sseg * 16;
      unsigned short* dst = &lds[srow * 256 + sseg * 16];
      async_cp16(src, dst);
      async_cp16(src + 8, dst + 8);
    }
    __builtin_amdgcn_s_wait_asynccnt(0);
#else
    if (sseg < ktn * 2) {
      const unsigned short* src = X + (size_t)srow * Ktot + (size_t)ktBase * 32 + sseg * 16;
      uint4 v0 = *(const uint4*)(src);
      uint4 v1 = *(const uint4*)(src + 8);
      *(uint4*)(&lds[srow * 256 + sseg * 16])     = v0;
      *(uint4*)(&lds[srow * 256 + sseg * 16 + 8]) = v1;
    }
#endif
    dma_publish(lds);         // escape + clobber: DMA writes to LDS are visible
    __syncthreads();

    if (ktBase + 8 < Kt) {  // prefetch next chunk's weight tiles into L2
      __builtin_prefetch(Wp + ((size_t)nt0 * Kt + (ktBase + 8)) * 512 + lane * 16, 0, 1);
      __builtin_prefetch(Wp + ((size_t)(nt0 + 1) * Kt + (ktBase + 8)) * 512 + lane * 16, 0, 1);
    }

    #pragma unroll 4
    for (int kt = 0; kt < ktn; ++kt) {
      const v16bf a  = *(const v16bf*)(&lds[(lane & 15) * 256 + kt * 32 + (lane >> 4) * 16]);
      const v16bf b0 = *(const v16bf*)(Wp + ((size_t)nt0 * Kt + (ktBase + kt)) * 512 + lane * 16);
      const v16bf b1 = *(const v16bf*)(Wp + ((size_t)(nt0 + 1) * Kt + (ktBase + kt)) * 512 + lane * 16);
      acc0 = __builtin_amdgcn_wmma_f32_16x16x32_bf16(false, a, false, b0, (short)0, acc0, false, false);
      acc1 = __builtin_amdgcn_wmma_f32_16x16x32_bf16(false, a, false, b1, (short)0, acc1, false, false);
    }
    __syncthreads();
  }

  // C/D layout: VGPR v, lanes 0-15 -> M=v, lanes 16-31 -> M=v+8; N=lane&15.
  const int colIn   = lane & 15;
  const int rowBase = (lane >> 4) << 3;
  const int col0 = nt0 * 16 + colIn;
  const int col1 = col0 + 16;
  const float bv0 = bias ? bias[col0] : 0.f;
  const float bv1 = bias ? bias[col1] : 0.f;
  #pragma unroll
  for (int v = 0; v < 8; ++v) {
    Y[(size_t)(rowBase + v) * ldY + col0] = acc0[v] + bv0;
    Y[(size_t)(rowBase + v) * ldY + col1] = acc1[v] + bv1;
  }
}

// ---------------------- per-step elementwise kernels -----------------------
__global__ void __launch_bounds__(256) prep_step(
    const float* __restrict__ stoch, float* __restrict__ deter,
    const int* __restrict__ action, const unsigned char* __restrict__ reset,
    const float* __restrict__ tokens, int t,
    unsigned short* __restrict__ x_in, unsigned short* __restrict__ x_gru,
    unsigned short* __restrict__ x_obs)
{
  const int b = blockIdx.x;
  const int tid = threadIdx.x;
  const bool rst = reset[b * LL + t] != 0;
  const float cf = rst ? 0.f : 1.f;

  for (int j = tid; j < SC; j += 256)
    x_in[b * 1056 + j] = f2bf(stoch[b * SC + j] * cf);

  const int act = rst ? 0 : action[b * LL + t];
  for (int j = tid; j < 32; j += 256) {
    float v = (!rst && j < ADIM && act == j) ? 1.f : 0.f;
    x_in[b * 1056 + 1024 + j] = f2bf(v);
  }

  for (int j = tid; j < DETER_D; j += 256) {
    float d = deter[b * DETER_D + j] * cf;
    deter[b * DETER_D + j] = d;
    x_gru[b * 5120 + 1024 + j] = f2bf(d);
  }

  for (int j = tid; j < TOKD; j += 256)
    x_obs[b * 8192 + j] = f2bf(tokens[((size_t)b * LL + t) * TOKD + j]);
}

// Stage one fp32 row into LDS via the async copy engine (ASYNCcnt path),
// then LN + silu reading LDS only (row otherwise read twice from HBM).
__global__ void __launch_bounds__(256) ln_silu_to_bf16(
    const float* __restrict__ X, const float* __restrict__ s,
    const float* __restrict__ o, int D,   // D <= 1024 at both call sites
    unsigned short* __restrict__ outB, int outLd, int outColOff)
{
  __shared__ __align__(16) float rowbuf[1024];
  __shared__ float r1[256], r2[256];
  const int row = blockIdx.x, tid = threadIdx.x;

#if defined(HAVE_ASYNC_LDS)
  for (int j = tid * 4; j < D; j += 1024)
    async_cp16(X + (size_t)row * D + j, rowbuf + j);
  __builtin_amdgcn_s_wait_asynccnt(0);
  dma_publish(rowbuf);
#else
  for (int j = tid; j < D; j += 256) rowbuf[j] = X[(size_t)row * D + j];
#endif
  __syncthreads();

  float a = 0.f, q = 0.f;
  for (int j = tid; j < D; j += 256) { float x = rowbuf[j]; a += x; q += x * x; }
  r1[tid] = a; r2[tid] = q; __syncthreads();
  for (int off = 128; off; off >>= 1) {
    if (tid < off) { r1[tid] += r1[tid + off]; r2[tid] += r2[tid + off]; }
    __syncthreads();
  }
  const float m  = r1[0] / D;
  const float vv = r2[0] / D - m * m;
  const float rs = rsqrtf(vv + 1e-5f);
  for (int j = tid; j < D; j += 256) {
    float y = (rowbuf[j] - m) * rs * s[j] + o[j];
    y = y / (1.f + expf(-y));          // silu
    outB[(size_t)row * outLd + outColOff + j] = f2bf(y);
  }
}

__global__ void __launch_bounds__(256) gru_gate(
    const float* __restrict__ G, const float* __restrict__ s,
    const float* __restrict__ o, float* __restrict__ deter,
    float* __restrict__ out_deter, unsigned short* __restrict__ x_obs, int t)
{
  __shared__ __align__(16) float rowbuf[3 * DETER_D];   // 48KB of 320KB LDS
  __shared__ float r1[256], r2[256];
  const int row = blockIdx.x, tid = threadIdx.x;
  const int D = 3 * DETER_D;

#if defined(HAVE_ASYNC_LDS)
  for (int j = tid * 4; j < D; j += 1024)
    async_cp16(G + (size_t)row * D + j, rowbuf + j);
  __builtin_amdgcn_s_wait_asynccnt(0);
  dma_publish(rowbuf);
#else
  for (int j = tid; j < D; j += 256) rowbuf[j] = G[(size_t)row * D + j];
#endif
  __syncthreads();

  float a = 0.f, q = 0.f;
  for (int j = tid; j < D; j += 256) { float x = rowbuf[j]; a += x; q += x * x; }
  r1[tid] = a; r2[tid] = q; __syncthreads();
  for (int off = 128; off; off >>= 1) {
    if (tid < off) { r1[tid] += r1[tid + off]; r2[tid] += r2[tid + off]; }
    __syncthreads();
  }
  const float m  = r1[0] / D;
  const float vv = r2[0] / D - m * m;
  const float rs = rsqrtf(vv + 1e-5f);
  for (int j = tid; j < DETER_D; j += 256) {
    float gr = (rowbuf[j]               - m) * rs * s[j]               + o[j];
    float gc = (rowbuf[DETER_D + j]     - m) * rs * s[DETER_D + j]     + o[DETER_D + j];
    float gu = (rowbuf[2 * DETER_D + j] - m) * rs * s[2 * DETER_D + j] + o[2 * DETER_D + j];
    float r  = 1.f / (1.f + expf(-gr));
    float c2 = tanhf(r * gc);
    float u  = 1.f / (1.f + expf(-(gu - 1.f)));
    float dn = u * c2 + (1.f - u) * deter[row * DETER_D + j];
    deter[row * DETER_D + j] = dn;
    out_deter[((size_t)row * LL + t) * DETER_D + j] = dn;
    x_obs[row * 8192 + 4096 + j] = f2bf(dn);
  }
}

// unimix softmax + Gumbel-argmax categorical; forward value of the
// straight-through sample is exactly the one-hot. Wave32 shfl reductions.
__global__ void __launch_bounds__(256) sample_kernel(
    const float* __restrict__ logits, int t,
    float* __restrict__ stoch, float* __restrict__ out_stoch)
{
  const int row  = blockIdx.x;
  const int wave = threadIdx.x >> 5;
  const int lane = threadIdx.x & 31;
  for (int i = 0; i < 4; ++i) {
    const int g = wave * 4 + i;
    const float lg = logits[((size_t)row * LL + t) * SC + g * 32 + lane];
    float mx = lg;
    for (int msk = 16; msk; msk >>= 1) mx = fmaxf(mx, __shfl_xor(mx, msk, 32));
    float e = expf(lg - mx);
    float ss = e;
    for (int msk = 16; msk; msk >>= 1) ss += __shfl_xor(ss, msk, 32);
    float p  = 0.99f * (e / ss) + 0.01f / 32.f;
    float ul = logf(p);
    unsigned h = hash4(0x243F6A88u + (unsigned)t, (unsigned)row, (unsigned)g, (unsigned)lane);
    float u01 = ((float)(h >> 8) + 0.5f) * (1.f / 16777216.f);
    float gmb = ul - logf(-logf(u01));
    float bv = gmb; int bi = lane;
    for (int msk = 16; msk; msk >>= 1) {
      float ov = __shfl_xor(bv, msk, 32);
      int   oi = __shfl_xor(bi, msk, 32);
      if (ov > bv || (ov == bv && oi < bi)) { bv = ov; bi = oi; }
    }
    float oh = (lane == bi) ? 1.f : 0.f;
    stoch[row * SC + g * 32 + lane] = oh;
    out_stoch[((size_t)row * LL + t) * SC + g * 32 + lane] = oh;
  }
}

// ------------------------------- driver ------------------------------------
extern "C" void kernel_launch(void* const* d_in, const int* in_sizes, int n_in,
                              void* d_out, int out_size, void* d_ws, size_t ws_size,
                              hipStream_t stream)
{
  const float* tokens = (const float*)d_in[0];
  const float* deter0 = (const float*)d_in[1];
  const float* stoch0 = (const float*)d_in[2];
  const float* w_in   = (const float*)d_in[3];
  const float* b_in   = (const float*)d_in[4];
  const float* s_in   = (const float*)d_in[5];
  const float* o_in   = (const float*)d_in[6];
  const float* w_gru  = (const float*)d_in[7];
  const float* b_gru  = (const float*)d_in[8];
  const float* s_gru  = (const float*)d_in[9];
  const float* o_gru  = (const float*)d_in[10];
  const float* w_obs  = (const float*)d_in[11];
  const float* b_obs  = (const float*)d_in[12];
  const float* s_obs  = (const float*)d_in[13];
  const float* o_obs  = (const float*)d_in[14];
  const float* w_post = (const float*)d_in[15];
  const float* b_post = (const float*)d_in[16];
  const int*   action = (const int*)d_in[17];
  const unsigned char* reset = (const unsigned char*)d_in[18];
  (void)in_sizes; (void)n_in; (void)out_size; (void)ws_size;

  char* ws = (char*)d_ws;
  size_t off = 0;
  auto alloc = [&](size_t bytes) -> char* {
    char* p = ws + off; off += (bytes + 255) & ~(size_t)255; return p;
  };
  unsigned short* WpIn   = (unsigned short*)alloc((size_t)KT_IN   * 64  * 512 * 2);
  unsigned short* WpGru  = (unsigned short*)alloc((size_t)KT_GRU  * 768 * 512 * 2);
  unsigned short* WpObs  = (unsigned short*)alloc((size_t)KT_OBS  * 64  * 512 * 2);
  unsigned short* WpPost = (unsigned short*)alloc((size_t)KT_POST * 64  * 512 * 2);
  unsigned short* Xin  = (unsigned short*)alloc((size_t)BB * 1056 * 2);
  unsigned short* Xgru = (unsigned short*)alloc((size_t)BB * 5120 * 2);
  unsigned short* Xobs = (unsigned short*)alloc((size_t)BB * 8192 * 2);
  unsigned short* Xpost= (unsigned short*)alloc((size_t)BB * 1024 * 2);
  float* Yin  = (float*)alloc((size_t)BB * 1024  * 4);
  float* Gbuf = (float*)alloc((size_t)BB * 12288 * 4);
  float* Yobs = (float*)alloc((size_t)BB * 1024  * 4);
  float* Dst  = (float*)alloc((size_t)BB * DETER_D * 4);
  float* Sst  = (float*)alloc((size_t)BB * SC * 4);

  float* out_stoch = (float*)d_out;
  float* out_deter = out_stoch + (size_t)BB * LL * SC;
  float* out_logit = out_deter + (size_t)BB * LL * DETER_D;

  (void)hipMemcpyAsync(Dst, deter0, (size_t)BB * DETER_D * sizeof(float),
                       hipMemcpyDeviceToDevice, stream);
  (void)hipMemcpyAsync(Sst, stoch0, (size_t)BB * SC * sizeof(float),
                       hipMemcpyDeviceToDevice, stream);

  auto packLaunch = [&](const float* W, unsigned short* Wp, int K, int Kt, int N) {
    size_t total = (size_t)Kt * (size_t)(N / 16) * 512;
    unsigned blocks = (unsigned)((total + 255) / 256);
    pack_weight<<<blocks, 256, 0, stream>>>(W, Wp, K, Kt, N);
  };
  packLaunch(w_in,   WpIn,   1030, KT_IN,   1024);
  packLaunch(w_gru,  WpGru,  5120, KT_GRU,  12288);
  packLaunch(w_obs,  WpObs,  8192, KT_OBS,  1024);
  packLaunch(w_post, WpPost, 1024, KT_POST, 1024);

  for (int t = 0; t < LL; ++t) {
    prep_step<<<BB, 256, 0, stream>>>(Sst, Dst, action, reset, tokens, t,
                                      Xin, Xgru, Xobs);
    gemm_bf16_wmma<<<1024 / 256, 256, 0, stream>>>(WpIn, Xin, b_in, Yin,
                                                   KT_IN, 1024, 1024);
    ln_silu_to_bf16<<<BB, 256, 0, stream>>>(Yin, s_in, o_in, 1024, Xgru, 5120, 0);
    gemm_bf16_wmma<<<12288 / 256, 256, 0, stream>>>(WpGru, Xgru, b_gru, Gbuf,
                                                    KT_GRU, 12288, 12288);
    gru_gate<<<BB, 256, 0, stream>>>(Gbuf, s_gru, o_gru, Dst, out_deter, Xobs, t);
    gemm_bf16_wmma<<<1024 / 256, 256, 0, stream>>>(WpObs, Xobs, b_obs, Yobs,
                                                   KT_OBS, 1024, 1024);
    ln_silu_to_bf16<<<BB, 256, 0, stream>>>(Yobs, s_obs, o_obs, 1024, Xpost, 1024, 0);
    gemm_bf16_wmma<<<1024 / 256, 256, 0, stream>>>(WpPost, Xpost, b_post,
                                                   out_logit + (size_t)t * SC,
                                                   KT_POST, 1024, LL * SC);
    sample_kernel<<<BB, 256, 0, stream>>>(out_logit, t, Sst, out_stoch);
  }
}
